// PointConv_19301583028465
// MI455X (gfx1250) — matrix operand order repaired and checked by reading, not compiled, not run on hardware
//
#include <hip/hip_runtime.h>
#include <hip/hip_bf16.h>
#include <hip/hip_fp16.h>

// ---------------- problem constants ----------------
#define BB   16
#define NN   4096
#define MM   1024
#define KNBR 32          // neighbors per query
#define CIN  64
#define C1   67          // CIN + 3 (use_xyz)
#define C1P  96          // padded to 3 * 32 for 16x16x32 WMMA K-chunks
#define H1   64
#define H2   128

typedef __attribute__((ext_vector_type(16))) _Float16 v16h;
typedef __attribute__((ext_vector_type(8)))  float    v8f;

union FragU { v16h v; unsigned u[8]; };

__device__ inline unsigned long long shfl_xor_u64(unsigned long long x, int off) {
  unsigned lo = (unsigned)x;
  unsigned hi = (unsigned)(x >> 32);
  lo = __shfl_xor(lo, off, 32);
  hi = __shfl_xor(hi, off, 32);
  return ((unsigned long long)hi << 32) | lo;
}

// =====================================================================
// Kernel 1: exact KNN (top-32 by squared distance), one wave32 per query.
// pos[b] staged in LDS (SoA, conflict-free lane striping).
// Per-lane register-resident sorted top-32 over a 128-point stripe,
// then 32-round cross-lane merge with packed (distBits,idx) u64 keys.
// =====================================================================
__global__ __launch_bounds__(128) void knn_kernel(
    const float* __restrict__ pos,   // [B, N, 3]
    const int*   __restrict__ idx,   // [B, M]
    int*         __restrict__ nn_out,   // [B, M, 32]
    float*       __restrict__ posq_out) // [B, M, 3]  (output segment 0)
{
  __shared__ float sx[NN];
  __shared__ float sy[NN];
  __shared__ float sz[NN];

  const int wg     = blockIdx.x;       // B * 256 workgroups
  const int b      = wg >> 8;
  const int wgInB  = wg & 255;
  const int tid    = threadIdx.x;

  const float* pb = pos + (size_t)b * NN * 3;
  for (int i = tid; i < NN; i += 128) {
    sx[i] = pb[i * 3 + 0];
    sy[i] = pb[i * 3 + 1];
    sz[i] = pb[i * 3 + 2];
  }
  __syncthreads();

  const int wave = tid >> 5;
  const int lane = tid & 31;
  const int m    = wgInB * 4 + wave;      // 4 queries per WG
  const int f    = b * MM + m;            // flat query id

  const int q = idx[f];
  const float qx = sx[q], qy = sy[q], qz = sz[q];

  // per-lane sorted ascending top-32 (registers only; all indices constant)
  float dl[32];
  int   il[32];
#pragma unroll
  for (int j = 0; j < 32; ++j) { dl[j] = 3.4e38f; il[j] = 0; }

  for (int t = 0; t < NN / 32; ++t) {
    const int i = t * 32 + lane;          // conflict-free LDS striping
    const float dx = sx[i] - qx;
    const float dy = sy[i] - qy;
    const float dz = sz[i] - qz;
    const float d  = dx * dx + dy * dy + dz * dz;
    if (d < dl[31]) {
      // branch-free unrolled sorted insert
#pragma unroll
      for (int j = 31; j >= 1; --j) {
        const bool shift = dl[j - 1] > d;
        const bool place = !shift && (dl[j] > d);
        if (shift)      { dl[j] = dl[j - 1]; il[j] = il[j - 1]; }
        else if (place) { dl[j] = d;         il[j] = i;         }
      }
      if (dl[0] > d) { dl[0] = d; il[0] = i; }
    }
  }

  // cross-lane merge: 32 rounds of wave argmin over each lane's head
  int myNN = 0;
  for (int k = 0; k < 32; ++k) {
    const unsigned db = __float_as_uint(dl[0]);  // distances >= 0: bit order == value order
    unsigned long long key  = ((unsigned long long)db << 32) | (unsigned)il[0];
    unsigned long long mine = key;
#pragma unroll
    for (int off = 16; off >= 1; off >>= 1) {
      const unsigned long long o = shfl_xor_u64(key, off);
      key = (o < key) ? o : key;
    }
    const int wi = (int)(key & 0xffffffffu);
    if (mine == key) {
      // winner pops its head (compile-time-index shift keeps arrays in VGPRs)
#pragma unroll
      for (int j = 0; j < 31; ++j) { dl[j] = dl[j + 1]; il[j] = il[j + 1]; }
      dl[31] = 3.4e38f;
    }
    if (lane == k) myNN = wi;
  }

  nn_out[(size_t)f * KNBR + lane] = myNN;
  if (lane < 3) {
    posq_out[(size_t)f * 3 + lane] = (lane == 0) ? qx : ((lane == 1) ? qy : qz);
  }
}

// =====================================================================
// Kernel 2: grouped 2-layer MLP via v_wmma_f32_16x16x32_f16 + neighbor max.
// 2 waves per WG; shared f16 weights (transposed, zero-padded) in LDS;
// per-wave x / h1 tile buffers in LDS. 56 WMMAs per query.
// =====================================================================
__global__ __launch_bounds__(64) void mlp_kernel(
    const float* __restrict__ pos,       // [B, N, 3]
    const float* __restrict__ feat,      // [B, N, 64]
    const float* __restrict__ W1,        // [67, 64]
    const float* __restrict__ b1v,       // [64]
    const float* __restrict__ W2,        // [64, 128]
    const float* __restrict__ b2v,       // [128]
    const float* __restrict__ posq,      // [B, M, 3] (from kernel 1)
    const int*   __restrict__ nn_in,     // [B, M, 32]
    float*       __restrict__ out)       // [B, M, 128] (output segment 1)
{
  __shared__ _Float16 sW1t[H1][C1P];     // [outCh][inCh padded]   12 KB
  __shared__ float    sB1[H1];
  __shared__ _Float16 sW2t[H2][H1];      // [outCh][inCh]          16 KB
  __shared__ float    sB2[H2];
  __shared__ _Float16 sX[2][KNBR][C1P];  // per-wave input tiles   12 KB
  __shared__ _Float16 sH[2][KNBR][H1];   // per-wave hidden tiles   8 KB

  const int tid  = threadIdx.x;
  const int wave = tid >> 5;
  const int lane = tid & 31;
  const int hi   = lane >> 4;            // lane half select (0/1)
  const int lm   = lane & 15;

  // ---- stage weights (transposed, f16, zero-padded K) ----
  for (int e = tid; e < H1 * C1P; e += 64) {
    const int c = e / C1P, r = e % C1P;
    sW1t[c][r] = (r < C1) ? (_Float16)W1[r * H1 + c] : (_Float16)0.0f;
  }
  for (int e = tid; e < H2 * H1; e += 64) {
    const int c = e / H1, r = e % H1;
    sW2t[c][r] = (_Float16)W2[r * H2 + c];
  }
  for (int e = tid; e < H1; e += 64) sB1[e] = b1v[e];
  for (int e = tid; e < H2; e += 64) sB2[e] = b2v[e];
  __syncthreads();

  // 8 queries per wave, 2 waves per WG, 1024 WGs -> 16384 queries
  for (int qi = 0; qi < 8; ++qi) {
    const int f = (blockIdx.x * 2 + wave) * 8 + qi;
    const int b = f >> 10;

    // ---- gather: row `lane` of x = [rel(3) | feat(64) | pad(29)] ----
    const int nid = nn_in[(size_t)f * KNBR + lane];
    const float pqx = posq[(size_t)f * 3 + 0];
    const float pqy = posq[(size_t)f * 3 + 1];
    const float pqz = posq[(size_t)f * 3 + 2];
    const float* pp = pos + ((size_t)b * NN + nid) * 3;
    sX[wave][lane][0] = (_Float16)(pp[0] - pqx);
    sX[wave][lane][1] = (_Float16)(pp[1] - pqy);
    sX[wave][lane][2] = (_Float16)(pp[2] - pqz);
    const float4* fp4 = (const float4*)(feat + ((size_t)b * NN + nid) * CIN);
#pragma unroll
    for (int c4 = 0; c4 < CIN / 4; ++c4) {
      const float4 v = fp4[c4];
      sX[wave][lane][3 + c4 * 4 + 0] = (_Float16)v.x;
      sX[wave][lane][3 + c4 * 4 + 1] = (_Float16)v.y;
      sX[wave][lane][3 + c4 * 4 + 2] = (_Float16)v.z;
      sX[wave][lane][3 + c4 * 4 + 3] = (_Float16)v.w;
    }
#pragma unroll
    for (int c = C1; c < C1P; ++c) sX[wave][lane][c] = (_Float16)0.0f;
    __syncthreads();

    // ---- layer 1: [32 x 96] @ [96 x 64] -> relu -> sH [32 x 64] ----
#pragma unroll
    for (int t = 0; t < 2; ++t) {
      const int arow = t * 16 + lm;         // A: lanes 0-15 and 16-31 both map rows 0-15
#pragma unroll
      for (int n = 0; n < 4; ++n) {
        v8f acc = {};
#pragma unroll
        for (int kc = 0; kc < 3; ++kc) {
          FragU a, w;
#pragma unroll
          for (int p = 0; p < 8; ++p) {
            const int g = p >> 2, pq = p & 3;
            const int Ka = kc * 32 + g * 16 + hi * 8 + pq * 2;   // A 16-bit layout
            a.u[p] = *(const unsigned*)&sX[wave][arow][Ka];
            const int Kb = kc * 32 + g * 16 + hi * 8 + pq * 2;   // B 16-bit layout
            w.u[p] = *(const unsigned*)&sW1t[n * 16 + lm][Kb];
          }
          acc = __builtin_amdgcn_wmma_f32_16x16x32_f16(
                    false, a.v, false, w.v, (short)0, acc, false, false);
        }
        const float bias = sB1[n * 16 + lm];
#pragma unroll
        for (int j = 0; j < 8; ++j) {
          float v = acc[j] + bias;
          v = v > 0.0f ? v : 0.0f;
          sH[wave][t * 16 + hi * 8 + j][n * 16 + lm] = (_Float16)v;  // D layout
        }
      }
    }
    __syncthreads();

    // ---- layer 2: [32 x 64] @ [64 x 128] -> relu -> max over 32 rows ----
    float cmax[8];
#pragma unroll
    for (int n = 0; n < 8; ++n) cmax[n] = 0.0f;   // relu output >= 0

#pragma unroll
    for (int t = 0; t < 2; ++t) {
      const int arow = t * 16 + lm;
#pragma unroll
      for (int n = 0; n < 8; ++n) {
        v8f acc = {};
#pragma unroll
        for (int kc = 0; kc < 2; ++kc) {
          FragU a, w;
#pragma unroll
          for (int p = 0; p < 8; ++p) {
            const int g = p >> 2, pq = p & 3;
            const int Ka = kc * 32 + g * 16 + hi * 8 + pq * 2;
            a.u[p] = *(const unsigned*)&sH[wave][arow][Ka];
            const int Kb = kc * 32 + g * 16 + hi * 8 + pq * 2;
            w.u[p] = *(const unsigned*)&sW2t[n * 16 + lm][Kb];
          }
          acc = __builtin_amdgcn_wmma_f32_16x16x32_f16(
                    false, a.v, false, w.v, (short)0, acc, false, false);
        }
        const float bias = sB2[n * 16 + lm];
#pragma unroll
        for (int j = 0; j < 8; ++j) {
          float v = acc[j] + bias;
          v = v > 0.0f ? v : 0.0f;
          cmax[n] = fmaxf(cmax[n], v);   // covers 16 of 32 rows for this lane's column
        }
      }
    }
    // fold the two lane halves (rows covered by lane^16), then write
#pragma unroll
    for (int n = 0; n < 8; ++n) {
      const float o = fmaxf(cmax[n], __shfl_xor(cmax[n], 16, 32));
      if (lane < 16) out[(size_t)f * H2 + n * 16 + lane] = o;
    }
    __syncthreads();  // before next query reuses sX/sH
  }
}

extern "C" void kernel_launch(void* const* d_in, const int* in_sizes, int n_in,
                              void* d_out, int out_size, void* d_ws, size_t ws_size,
                              hipStream_t stream) {
  const float* pos  = (const float*)d_in[0];   // [16,4096,3]
  const float* feat = (const float*)d_in[1];   // [16,4096,64]
  const float* W1   = (const float*)d_in[2];   // [67,64]
  const float* b1   = (const float*)d_in[3];   // [64]
  const float* W2   = (const float*)d_in[4];   // [64,128]
  const float* b2   = (const float*)d_in[5];   // [128]
  const int*   idx  = (const int*)d_in[6];     // [16,1024]

  float* posq_out = (float*)d_out;                       // [16,1024,3]
  float* out2     = (float*)d_out + (size_t)BB * MM * 3; // [16,1024,128]
  int*   nn       = (int*)d_ws;                          // [16,1024,32] = 2 MB scratch

  // one wave per query, 4 queries per 128-thread WG, pos[b] staged per WG
  knn_kernel<<<BB * (MM / 4), 128, 0, stream>>>(pos, idx, nn, posq_out);

  // 2 waves per 64-thread WG, 8 queries per wave
  mlp_kernel<<<(BB * MM) / 16, 64, 0, stream>>>(pos, feat, W1, b1, W2, b2,
                                                posq_out, nn, out2);
}
